// MoEsparseRouting_33251636806445
// MI455X (gfx1250) — matrix operand order repaired and checked by reading, not compiled, not run on hardware
//
#include <hip/hip_runtime.h>
#include <stdint.h>
#include <stddef.h>

// ---------------------------------------------------------------- types
typedef _Float16 h16v __attribute__((ext_vector_type(16)));
typedef _Float16 h8v  __attribute__((ext_vector_type(8)));
typedef float    f8v  __attribute__((ext_vector_type(8)));

union H16 { h16v v; h8v h[2]; _Float16 e[16]; };

constexpr int kB  = 32;
constexpr int kS  = 512;
constexpr int kD  = 768;
constexpr int kH  = 12;
constexpr int kHD = 64;
constexpr int kL  = 12;
constexpr int kE  = 8;
constexpr int kM  = kB * kS;          // 16384 tokens
constexpr float kAlpha = 8.0f;

__device__ __forceinline__ f8v wmma_f16(h16v a, h16v b, f8v c) {
  // v_wmma_f32_16x16x32_f16  D = A(16x32) * B(32x16) + C
  return __builtin_amdgcn_wmma_f32_16x16x32_f16(false, a, false, b, (short)0, c,
                                                false, false);
}

// CDNA5 async memory->LDS copy (ASYNCcnt-tracked).  LDS address = low 32 bits
// of the generic pointer.
__device__ __forceinline__ void async_ld_b128(const void* lds_ptr, const void* gptr) {
  unsigned lds = (unsigned)(uintptr_t)lds_ptr;
  unsigned long long ga = (unsigned long long)(uintptr_t)gptr;
  asm volatile("global_load_async_to_lds_b128 %0, %1, off"
               :: "v"(lds), "v"(ga) : "memory");
}
__device__ __forceinline__ void async_wait0() {
  asm volatile("s_wait_asynccnt 0x0" ::: "memory");
}

// ---------------------------------------------------------------- router
// expert = argmax_e ( mean_s(hidden[B-1,s,:]) @ router_weights )[e]
__global__ __launch_bounds__(256) void router_kernel(
    const float* __restrict__ hs, const float* __restrict__ rw, int* eidx) {
  __shared__ float pooled[kD];
  __shared__ float logits[kE];
  const int t = threadIdx.x;
  for (int d = t; d < kD; d += 256) {
    float s = 0.f;
    const float* p = hs + (size_t)(kB - 1) * kS * kD + d;
    for (int si = 0; si < kS; ++si) s += p[(size_t)si * kD];
    pooled[d] = s * (1.0f / kS);
  }
  __syncthreads();
  if (t < kE) {
    float acc = 0.f;
    for (int d = 0; d < kD; ++d) acc += pooled[d] * rw[d * kE + t];
    logits[t] = acc;
  }
  __syncthreads();
  if (t == 0) {
    int best = 0; float bv = logits[0];
    for (int e = 1; e < kE; ++e)
      if (logits[e] > bv) { bv = logits[e]; best = e; }
    *eidx = best;
  }
}

// ---------------------------------------------------------------- f32 -> f16
__global__ __launch_bounds__(256) void cvt_kernel(const float* __restrict__ src,
                                                  _Float16* __restrict__ dst,
                                                  size_t n) {
  size_t i = (size_t)blockIdx.x * blockDim.x + threadIdx.x;
  size_t stride = (size_t)gridDim.x * blockDim.x;
  for (; i < n; i += stride) dst[i] = (_Float16)src[i];
}

__global__ __launch_bounds__(256) void initx_kernel(const float* __restrict__ src,
                                                    float* __restrict__ xf,
                                                    _Float16* __restrict__ xh,
                                                    size_t n) {
  size_t i = (size_t)blockIdx.x * blockDim.x + threadIdx.x;
  size_t stride = (size_t)gridDim.x * blockDim.x;
  for (; i < n; i += stride) { float v = src[i]; xf[i] = v; xh[i] = (_Float16)v; }
}

// ---------------------------------------------------------------- TT adapt
// W_a[l] = f16( W[l] + 8 * Left(768x8) @ Right(8x768) )
// Left[j,a,n,o]  = sum_{k,m} c0[j,k] c1[k,a,m] c2[m,n,o]
// Right[o,p,r,t] = sum_{q,s} c3[o,p,q] c4[q,r,s] c5[s,t]
__global__ __launch_bounds__(256) void tt_adapt_kernel(
    const float* __restrict__ c0, const float* __restrict__ c1,
    const float* __restrict__ c2, const float* __restrict__ c3,
    const float* __restrict__ c4, const float* __restrict__ c5,
    const float* __restrict__ Wbase, _Float16* __restrict__ Wout,
    const int* __restrict__ eidx) {
  __shared__ float sT1[768];     // [j][a][m]
  __shared__ float sLeft[6144];  // [j][a][n][o] ; row = j*64+a*8+n
  const int layer = blockIdx.x;
  const int tid = threadIdx.x;
  const int e = *eidx;
  const float* C0 = c0 + ((size_t)e * kL + layer) * 96;    // [12][8]
  const float* C1 = c1 + ((size_t)e * kL + layer) * 512;   // [k][a][m]
  const float* C2 = c2 + ((size_t)e * kL + layer) * 512;   // [m][n][o]
  const float* C3 = c3 + ((size_t)e * kL + layer) * 512;   // [o][p][q]
  const float* C4 = c4 + ((size_t)e * kL + layer) * 512;   // [q][r][s]
  const float* C5 = c5 + ((size_t)e * kL + layer) * 96;    // [s][t]

  for (int idx = tid; idx < 768; idx += 256) {
    const int m = idx & 7, a = (idx >> 3) & 7, j = idx >> 6;
    float s = 0.f;
#pragma unroll
    for (int k = 0; k < 8; ++k) s += C0[j * 8 + k] * C1[(k * 8 + a) * 8 + m];
    sT1[idx] = s;
  }
  __syncthreads();
  for (int idx = tid; idx < 6144; idx += 256) {
    const int o = idx & 7, n = (idx >> 3) & 7, a = (idx >> 6) & 7, j = idx >> 9;
    float s = 0.f;
#pragma unroll
    for (int m = 0; m < 8; ++m) s += sT1[(j * 8 + a) * 8 + m] * C2[(m * 8 + n) * 8 + o];
    sLeft[idx] = s;
  }
  __syncthreads();

  // each thread owns 3 columns: col = c*256 + tid
  float R[3][8];
#pragma unroll
  for (int c = 0; c < 3; ++c) {
    const int col = c * 256 + tid;
    const int pp = col / 96, rr = (col / 12) % 8, t2 = col % 12;
#pragma unroll
    for (int o = 0; o < 8; ++o) {
      float acc = 0.f;
#pragma unroll
      for (int s = 0; s < 8; ++s) {
        float aq = 0.f;
#pragma unroll
        for (int q = 0; q < 8; ++q)
          aq += C3[(o * 8 + pp) * 8 + q] * C4[(q * 8 + rr) * 8 + s];
        acc += aq * C5[s * 12 + t2];
      }
      R[c][o] = acc;
    }
  }
  const float* Wl = Wbase + (size_t)layer * kD * kD;
  _Float16* Wo = Wout + (size_t)layer * kD * kD;
  for (int row = 0; row < kD; ++row) {
    const float* lf = &sLeft[row * 8];
    float l0 = lf[0], l1 = lf[1], l2 = lf[2], l3 = lf[3];
    float l4 = lf[4], l5 = lf[5], l6 = lf[6], l7 = lf[7];
#pragma unroll
    for (int c = 0; c < 3; ++c) {
      const int col = c * 256 + tid;
      float acc = l0 * R[c][0] + l1 * R[c][1] + l2 * R[c][2] + l3 * R[c][3] +
                  l4 * R[c][4] + l5 * R[c][5] + l6 * R[c][6] + l7 * R[c][7];
      const size_t w = (size_t)row * kD + col;
      Wo[w] = (_Float16)(Wl[w] + kAlpha * acc);
    }
  }
}

// ---------------------------------------------------------------- WMMA GEMM
// C[m,n] = sum_k A[m,k] * W[n,k]  (+bias[n]) ; M=16384, N=K=768
// mode 0: outh = f16(C) ; mode 1: outf[m,n] += C (residual in place)
// block tile 64x256 (8 waves, wave tile 32x64); A staged in LDS via
// global_load_async_to_lds_b128; all 4 B fragments preloaded per K step so
// the 8 WMMAs issue back-to-back while the next loads are in flight.
__global__ __launch_bounds__(256) void wmma_gemm_kernel(
    const _Float16* __restrict__ A, const _Float16* __restrict__ W,
    const float* __restrict__ bias, _Float16* __restrict__ outh,
    float* __restrict__ outf, int mode) {
  __shared__ _Float16 sA[64][32];
  const int tid = threadIdx.x;
  const int lane = tid & 31, wid = tid >> 5;
  const int ll = lane & 15, lh = lane >> 4;
  const int wm = wid & 1, wn = wid >> 1;
  const int blockM = blockIdx.x * 64;
  const int blockN = blockIdx.y * 256;

  f8v acc[2][4];
  {
    f8v z = {};
#pragma unroll
    for (int i = 0; i < 2; ++i)
#pragma unroll
      for (int j = 0; j < 4; ++j) acc[i][j] = z;
  }
  const int ldr = tid >> 2;            // 0..63
  const int ldc = (tid & 3) * 8;       // 0,8,16,24
  const _Float16* aRow = &A[(size_t)(blockM + ldr) * kD + ldc];

  // per-fragment W row pointers (advance by 32 halves per K step)
  const _Float16* wp[4];
#pragma unroll
  for (int nf = 0; nf < 4; ++nf)
    wp[nf] = &W[(size_t)(blockN + wn * 64 + nf * 16 + ll) * kD + lh * 8];

  for (int k0 = 0; k0 < kD; k0 += 32) {
    __syncthreads();                         // prev-iter readers done
    async_ld_b128(&sA[ldr][ldc], aRow + k0); // 16B memory -> LDS, ASYNCcnt
    async_wait0();
    __syncthreads();                         // tile published

    H16 af[2];
#pragma unroll
    for (int mf = 0; mf < 2; ++mf) {
      const int m = wm * 32 + mf * 16 + ll;
      af[mf].h[0] = *(const h8v*)&sA[m][lh * 8];
      af[mf].h[1] = *(const h8v*)&sA[m][lh * 8 + 16];
    }
    H16 bf[4];
#pragma unroll
    for (int nf = 0; nf < 4; ++nf) {
      bf[nf].h[0] = *(const h8v*)(wp[nf]);
      bf[nf].h[1] = *(const h8v*)(wp[nf] + 16);
      __builtin_prefetch((const void*)(wp[nf] + 32), 0, 1);  // next K tile
      wp[nf] += 32;
    }
#pragma unroll
    for (int nf = 0; nf < 4; ++nf) {
      acc[0][nf] = wmma_f16(af[0].v, bf[nf].v, acc[0][nf]);
      acc[1][nf] = wmma_f16(af[1].v, bf[nf].v, acc[1][nf]);
    }
  }
#pragma unroll
  for (int mf = 0; mf < 2; ++mf)
#pragma unroll
    for (int nf = 0; nf < 4; ++nf) {
      const int ncol = blockN + wn * 64 + nf * 16 + ll;
      const float bb = bias[ncol];
      const int rbase = blockM + wm * 32 + mf * 16 + 8 * lh;
#pragma unroll
      for (int i = 0; i < 8; ++i) {
        const size_t o = (size_t)(rbase + i) * kD + ncol;
        const float v = acc[mf][nf][i] + bb;
        if (mode == 0) outh[o] = (_Float16)v;
        else           outf[o] += v;
      }
    }
}

// ---------------------------------------------------------------- attention
// one block = (b, h, 128-query strip); 8 waves x 16 query rows; flash softmax
__global__ __launch_bounds__(256) void attn_kernel(
    const _Float16* __restrict__ qb, const _Float16* __restrict__ kb,
    const _Float16* __restrict__ vb, _Float16* __restrict__ ctxb) {
  __shared__ _Float16 sK[64][64];
  __shared__ _Float16 sV[64][64];
  __shared__ _Float16 sP[8][16][32];
  const int tid = threadIdx.x;
  const int lane = tid & 31, wid = tid >> 5;
  const int ll = lane & 15, lh = lane >> 4;
  const int bh = blockIdx.x >> 2;
  const int qblk = blockIdx.x & 3;
  const int b = bh / kH, h = bh % kH;
  const int qRow0 = qblk * 128 + wid * 16;
  const size_t rowBase = (size_t)b * kS;
  const float scale = 0.125f;  // 1/sqrt(64)

  // Q fragments (hd = 64 -> two K=32 fragments)
  H16 aq[2];
  {
    const int m = qRow0 + ll;
    const _Float16* qp = qb + (rowBase + m) * kD + h * kHD;
#pragma unroll
    for (int j = 0; j < 2; ++j) {
      aq[j].h[0] = *(const h8v*)(qp + j * 32 + lh * 8);
      aq[j].h[1] = *(const h8v*)(qp + j * 32 + lh * 8 + 16);
    }
  }
  f8v ctx[4];
  {
    f8v z = {};
#pragma unroll
    for (int n = 0; n < 4; ++n) ctx[n] = z;
  }
  float mrow[8], lrow[8];
#pragma unroll
  for (int i = 0; i < 8; ++i) { mrow[i] = -1e30f; lrow[i] = 0.f; }

  const int ldr = tid >> 2;            // 0..63
  const int ldc = (tid & 3) * 16;      // 0,16,32,48

  for (int kc = 0; kc < kS; kc += 64) {
    __syncthreads();
    {
      const _Float16* kp = kb + (rowBase + kc + ldr) * kD + h * kHD + ldc;
      const _Float16* vp = vb + (rowBase + kc + ldr) * kD + h * kHD + ldc;
      async_ld_b128(&sK[ldr][ldc],     kp);      // 16B + 16B per matrix
      async_ld_b128(&sK[ldr][ldc + 8], kp + 8);
      async_ld_b128(&sV[ldr][ldc],     vp);
      async_ld_b128(&sV[ldr][ldc + 8], vp + 8);
      async_wait0();
    }
    __syncthreads();

    // scores: 16 query rows x 64 keys
    f8v sc[4];
#pragma unroll
    for (int nt = 0; nt < 4; ++nt) {
      const int key = nt * 16 + ll;
      H16 bk0, bk1;
      bk0.h[0] = *(const h8v*)&sK[key][lh * 8];
      bk0.h[1] = *(const h8v*)&sK[key][lh * 8 + 16];
      bk1.h[0] = *(const h8v*)&sK[key][32 + lh * 8];
      bk1.h[1] = *(const h8v*)&sK[key][32 + lh * 8 + 16];
      f8v c = {};
      c = wmma_f16(aq[0].v, bk0.v, c);
      c = wmma_f16(aq[1].v, bk1.v, c);
#pragma unroll
      for (int i = 0; i < 8; ++i) c[i] *= scale;
      sc[nt] = c;
    }

    // online softmax update (row reductions across 16-lane halves)
    float corr[8];
#pragma unroll
    for (int i = 0; i < 8; ++i) {
      float t = fmaxf(fmaxf(sc[0][i], sc[1][i]), fmaxf(sc[2][i], sc[3][i]));
#pragma unroll
      for (int off = 1; off < 16; off <<= 1) t = fmaxf(t, __shfl_xor(t, off, 16));
      const float mnew = fmaxf(mrow[i], t);
      corr[i] = __expf(mrow[i] - mnew);
      mrow[i] = mnew;
    }
#pragma unroll
    for (int nt = 0; nt < 4; ++nt)
#pragma unroll
      for (int i = 0; i < 8; ++i) sc[nt][i] = __expf(sc[nt][i] - mrow[i]);
#pragma unroll
    for (int i = 0; i < 8; ++i) {
      float rs = sc[0][i] + sc[1][i] + sc[2][i] + sc[3][i];
#pragma unroll
      for (int off = 1; off < 16; off <<= 1) rs += __shfl_xor(rs, off, 16);
      lrow[i] = lrow[i] * corr[i] + rs;
#pragma unroll
      for (int n = 0; n < 4; ++n) ctx[n][i] *= corr[i];
    }

    // P @ V (two 32-key halves); P re-layout C-frag -> A-frag via LDS
#pragma unroll
    for (int hh = 0; hh < 2; ++hh) {
      __syncthreads();
#pragma unroll
      for (int fr = 0; fr < 2; ++fr)
#pragma unroll
        for (int i = 0; i < 8; ++i)
          sP[wid][8 * lh + i][fr * 16 + ll] = (_Float16)sc[hh * 2 + fr][i];
      __syncthreads();
      H16 pa;
      pa.h[0] = *(const h8v*)&sP[wid][ll][lh * 8];
      pa.h[1] = *(const h8v*)&sP[wid][ll][lh * 8 + 16];
#pragma unroll
      for (int n = 0; n < 4; ++n) {
        H16 bvv;
        const int hdc = n * 16 + ll;
        const int kbase = hh * 32 + lh * 8;
#pragma unroll
        for (int kk = 0; kk < 16; ++kk) bvv.e[kk] = sV[kbase + kk + (kk >= 8 ? 8 : 0)][hdc];
        ctx[n] = wmma_f16(pa.v, bvv.v, ctx[n]);
      }
    }
  }

  // epilogue
#pragma unroll
  for (int n = 0; n < 4; ++n) {
    const int hdc = h * kHD + n * 16 + ll;
#pragma unroll
    for (int i = 0; i < 8; ++i) {
      const int row = qRow0 + 8 * lh + i;
      ctxb[(rowBase + row) * kD + hdc] = (_Float16)(ctx[n][i] / lrow[i]);
    }
  }
}

// ---------------------------------------------------------------- layernorm
__global__ __launch_bounds__(256) void ln_kernel(
    const float* __restrict__ xin, const float* __restrict__ gamma,
    const float* __restrict__ beta, float* __restrict__ dstf,
    _Float16* __restrict__ dsth) {
  __shared__ float red[256];
  const int t = threadIdx.x;
  const size_t row = blockIdx.x;
  const float* xr = xin + row * kD;
  float v0 = xr[t], v1 = xr[t + 256], v2 = xr[t + 512];

  red[t] = v0 + v1 + v2;
  __syncthreads();
  for (int s = 128; s > 0; s >>= 1) {
    if (t < s) red[t] += red[t + s];
    __syncthreads();
  }
  const float mean = red[0] * (1.0f / kD);
  __syncthreads();
  const float d0 = v0 - mean, d1 = v1 - mean, d2 = v2 - mean;
  red[t] = d0 * d0 + d1 * d1 + d2 * d2;
  __syncthreads();
  for (int s = 128; s > 0; s >>= 1) {
    if (t < s) red[t] += red[t + s];
    __syncthreads();
  }
  const float inv = rsqrtf(red[0] * (1.0f / kD) + 1e-12f);
  const float y0 = d0 * inv * gamma[t]       + beta[t];
  const float y1 = d1 * inv * gamma[t + 256] + beta[t + 256];
  const float y2 = d2 * inv * gamma[t + 512] + beta[t + 512];
  float* df = dstf + row * kD;
  _Float16* dh = dsth + row * kD;
  df[t] = y0;       dh[t] = (_Float16)y0;
  df[t + 256] = y1; dh[t + 256] = (_Float16)y1;
  df[t + 512] = y2; dh[t + 512] = (_Float16)y2;
}

// ---------------------------------------------------------------- host
extern "C" void kernel_launch(void* const* d_in, const int* in_sizes, int n_in,
                              void* d_out, int out_size, void* d_ws, size_t ws_size,
                              hipStream_t stream) {
  (void)in_sizes; (void)n_in; (void)out_size; (void)ws_size;
  const float* hs = (const float*)d_in[0];
  const float* rw = (const float*)d_in[1];
  const float* qc[6], *vc[6];
  for (int i = 0; i < 6; ++i) { qc[i] = (const float*)d_in[2 + i]; vc[i] = (const float*)d_in[8 + i]; }
  const float* Wq = (const float*)d_in[14];
  const float* Wk = (const float*)d_in[15];
  const float* Wv = (const float*)d_in[16];
  const float* Wo = (const float*)d_in[17];
  const float* bq = (const float*)d_in[18];
  const float* bk = (const float*)d_in[19];
  const float* bv = (const float*)d_in[20];
  const float* bo = (const float*)d_in[21];
  const float* lns = (const float*)d_in[22];
  const float* lnb = (const float*)d_in[23];

  // workspace carve-out
  size_t off = 0;
  auto carve = [&](size_t bytes) -> void* {
    off = (off + 255) & ~(size_t)255;
    void* p = (char*)d_ws + off;
    off += bytes;
    return p;
  };
  const size_t wElems = (size_t)kL * kD * kD;
  const size_t aElems = (size_t)kM * kD;
  int*       eidx = (int*)carve(sizeof(int));
  _Float16*  WqA  = (_Float16*)carve(wElems * 2);
  _Float16*  WvA  = (_Float16*)carve(wElems * 2);
  _Float16*  WkH  = (_Float16*)carve(wElems * 2);
  _Float16*  WoH  = (_Float16*)carve(wElems * 2);
  float*     xf   = (float*)carve(aElems * 4);
  _Float16*  xh   = (_Float16*)carve(aElems * 2);
  _Float16*  qh   = (_Float16*)carve(aElems * 2);
  _Float16*  kh   = (_Float16*)carve(aElems * 2);
  _Float16*  vh   = (_Float16*)carve(aElems * 2);
  _Float16*  ctxh = (_Float16*)carve(aElems * 2);

  router_kernel<<<1, 256, 0, stream>>>(hs, rw, eidx);
  cvt_kernel<<<4096, 256, 0, stream>>>(Wk, WkH, wElems);
  cvt_kernel<<<4096, 256, 0, stream>>>(Wo, WoH, wElems);
  initx_kernel<<<4096, 256, 0, stream>>>(hs, xf, xh, aElems);
  tt_adapt_kernel<<<kL, 256, 0, stream>>>(qc[0], qc[1], qc[2], qc[3], qc[4], qc[5],
                                          Wq, WqA, eidx);
  tt_adapt_kernel<<<kL, 256, 0, stream>>>(vc[0], vc[1], vc[2], vc[3], vc[4], vc[5],
                                          Wv, WvA, eidx);

  const dim3 gemmGrid(kM / 64, kD / 256);
  for (int l = 0; l < kL; ++l) {
    const size_t wOff = (size_t)l * kD * kD;
    wmma_gemm_kernel<<<gemmGrid, 256, 0, stream>>>(xh, WqA + wOff, bq + l * kD,
                                                   qh, nullptr, 0);
    wmma_gemm_kernel<<<gemmGrid, 256, 0, stream>>>(xh, WkH + wOff, bk + l * kD,
                                                   kh, nullptr, 0);
    wmma_gemm_kernel<<<gemmGrid, 256, 0, stream>>>(xh, WvA + wOff, bv + l * kD,
                                                   vh, nullptr, 0);
    attn_kernel<<<kB * kH * (kS / 128), 256, 0, stream>>>(qh, kh, vh, ctxh);
    // out-projection with residual accumulated in place into xf
    wmma_gemm_kernel<<<gemmGrid, 256, 0, stream>>>(ctxh, WoH + wOff, bo + l * kD,
                                                   nullptr, xf, 1);
    ln_kernel<<<kM, 256, 0, stream>>>(xf, lns + l * kD, lnb + l * kD,
                                      (l == kL - 1) ? (float*)d_out : xf, xh);
  }
}
